// MixtureOfExperts_64476049047588
// MI455X (gfx1250) — compile-verified, matrix operand order checked by reading, not run on hardware
//
#include <hip/hip_runtime.h>
#include <math.h>

// ---------------- problem constants ----------------
constexpr int NTOK = 8192;   // B*T
constexpr int DDIM = 2048;   // model dim
constexpr int FDIM = 5632;   // ffn dim
constexpr int NEXP = 8;
constexpr int RBLK = NTOK / 8;   // router blocks (8 waves / block, 1 token per wave)

// ---------------- WMMA types ----------------
typedef __bf16 bf16;
typedef __attribute__((ext_vector_type(16))) __bf16 v16bf;
typedef __attribute__((ext_vector_type(8)))  __bf16 v8bf;
typedef __attribute__((ext_vector_type(8)))  float  v8f;

constexpr int LDK = 40;  // LDS row stride in bf16 (32 data + 8 pad; 80B => bank-spread, 16B aligned)

// ---------------- helpers ----------------
// f32 -> bf16 (round to nearest, ties away): bits + 0x8000, take high half.
__device__ inline bf16 f2bf(float f) {
  unsigned u = __builtin_bit_cast(unsigned, f) + 0x8000u;
  union { unsigned short s; bf16 b; } o; o.s = (unsigned short)(u >> 16); return o.b;
}
// packed f32x2 -> bf16x2: v_cvt_pk_bf16_f32 when declared, else 2x v_add + 1x v_perm_b32
__device__ inline unsigned pk2(float a, float b) {
#if __has_builtin(__builtin_amdgcn_cvt_pk_bf16_f32)
  return __builtin_bit_cast(unsigned, __builtin_amdgcn_cvt_pk_bf16_f32(a, b));
#else
  unsigned ua = __builtin_bit_cast(unsigned, a) + 0x8000u;
  unsigned ub = __builtin_bit_cast(unsigned, b) + 0x8000u;
  return __builtin_amdgcn_perm(ub, ua, 0x07060302u);  // bytes {b3,b2,a3,a2} = {b.hi16, a.hi16}
#endif
}
__device__ inline uint4 pack8(float4 a, float4 b) {
  uint4 o;
  o.x = pk2(a.x, a.y);
  o.y = pk2(a.z, a.w);
  o.z = pk2(b.x, b.y);
  o.w = pk2(b.z, b.w);
  return o;
}

// ---- CDNA5 async global->LDS copy (ASYNCcnt-tracked, bypasses VGPRs) ----
// INST_OFFSET applies to both the LDS and global sides, so one address pair +
// offset:16 moves this thread's 32 contiguous bytes.
__device__ inline unsigned lds_off_of(const void* p) {
  return (unsigned)(uintptr_t)p;   // low 32 bits of flat shared address == LDS offset
}
__device__ inline void async_ld_b128x2(unsigned lds_off, const void* g) {
  asm volatile("global_load_async_to_lds_b128 %0, %1, off\n\t"
               "global_load_async_to_lds_b128 %0, %1, off offset:16"
               :: "v"(lds_off), "v"(g) : "memory");
}
__device__ inline void wait_async0() {
  asm volatile("s_wait_asynccnt 0" ::: "memory");
}

// ISA 16-bit operand layout (16x32): lanes 0-15 hold K{0..7,16..23}, lanes 16-31 hold K{8..15,24..31}.
__device__ inline v16bf load_frag(const bf16* rowp, int half) {
  v8bf lo = *(const v8bf*)(rowp + half * 8);        // K 0..7  (or 8..15)
  v8bf hi = *(const v8bf*)(rowp + half * 8 + 16);   // K 16..23 (or 24..31)
  return __builtin_shufflevector(lo, hi, 0,1,2,3,4,5,6,7,8,9,10,11,12,13,14,15);
}

// ---------------- utility kernels ----------------
__global__ void zero_f4(float4* p, int n4) {
  int i = blockIdx.x * blockDim.x + threadIdx.x;
  if (i < n4) p[i] = make_float4(0.f, 0.f, 0.f, 0.f);
}
__global__ void zero_u(unsigned* p, int n) {
  int i = blockIdx.x * blockDim.x + threadIdx.x;
  if (i < n) p[i] = 0u;
}
__global__ void cvt_x(const float* __restrict__ x, bf16* __restrict__ xb, int n4) {
  int i = blockIdx.x * blockDim.x + threadIdx.x;
  if (i < n4) {
    float4 v = ((const float4*)x)[i];
    uint2 o;
    o.x = pk2(v.x, v.y);
    o.y = pk2(v.z, v.w);
    ((uint2*)xb)[i] = o;
  }
}

// ---------------- router: wave32 per token ----------------
__global__ __launch_bounds__(256) void router_kernel(
    const float* __restrict__ x, const float* __restrict__ rw,
    float* __restrict__ combine, unsigned* __restrict__ counts,
    int* __restrict__ lists, float* __restrict__ score_part)
{
  __shared__ float sScore[8][NEXP];  // per-wave softmax scores for deterministic block partials
  int wave = threadIdx.x >> 5, lane = threadIdx.x & 31;
  int n = blockIdx.x * 8 + wave;
  const float* xr = x + (size_t)n * DDIM;
  float acc[NEXP];
#pragma unroll
  for (int e = 0; e < NEXP; ++e) acc[e] = 0.f;
  for (int j = lane; j < DDIM; j += 32) {
    float xv = xr[j];
#pragma unroll
    for (int e = 0; e < NEXP; ++e) acc[e] += xv * rw[(size_t)e * DDIM + j];
  }
#pragma unroll
  for (int e = 0; e < NEXP; ++e)
#pragma unroll
    for (int off = 16; off > 0; off >>= 1) acc[e] += __shfl_xor(acc[e], off, 32);

  if (lane == 0) {
    float mx = acc[0];
#pragma unroll
    for (int e = 1; e < NEXP; ++e) mx = fmaxf(mx, acc[e]);
    float ex[NEXP], s = 0.f;
#pragma unroll
    for (int e = 0; e < NEXP; ++e) { ex[e] = __expf(acc[e] - mx); s += ex[e]; }
    float sc[NEXP];
#pragma unroll
    for (int e = 0; e < NEXP; ++e) sc[e] = ex[e] / s;
    int e0 = 0;
#pragma unroll
    for (int e = 1; e < NEXP; ++e) if (sc[e] > sc[e0]) e0 = e;   // ties -> lowest index
    int e1 = (e0 == 0) ? 1 : 0;
#pragma unroll
    for (int e = 0; e < NEXP; ++e) if (e != e0 && e != e1 && sc[e] > sc[e1]) e1 = e;
    float w0 = sc[e0], w1 = sc[e1], wsum = w0 + w1;
    w0 /= wsum; w1 /= wsum;
#pragma unroll
    for (int e = 0; e < NEXP; ++e)
      combine[(size_t)n * NEXP + e] = (e == e0) ? w0 : (e == e1) ? w1 : 0.f;
    unsigned p0 = atomicAdd(&counts[e0], 1u); lists[(size_t)e0 * NTOK + p0] = n;
    unsigned p1 = atomicAdd(&counts[e1], 1u); lists[(size_t)e1 * NTOK + p1] = n;
#pragma unroll
    for (int e = 0; e < NEXP; ++e) sScore[wave][e] = sc[e];
  }
  __syncthreads();
  if (threadIdx.x == 0) {
#pragma unroll
    for (int e = 0; e < NEXP; ++e) {
      float s = 0.f;
      for (int w = 0; w < 8; ++w) s += sScore[w][e];  // fixed order -> deterministic
      score_part[(size_t)blockIdx.x * NEXP + e] = s;
    }
  }
}

__global__ void aux_finalize(const float* __restrict__ score_part,
                             const unsigned* __restrict__ counts,
                             float* __restrict__ aux_out, int nblocks)
{
  if (blockIdx.x == 0 && threadIdx.x == 0) {
    float s[NEXP];
#pragma unroll
    for (int e = 0; e < NEXP; ++e) s[e] = 0.f;
    for (int b = 0; b < nblocks; ++b)
#pragma unroll
      for (int e = 0; e < NEXP; ++e) s[e] += score_part[(size_t)b * NEXP + e];
    float aux = 0.f;
#pragma unroll
    for (int e = 0; e < NEXP; ++e)
      aux += (s[e] / (float)NTOK) * ((float)counts[e] / (float)NTOK);
    aux_out[0] = (float)NEXP * aux;
  }
}

// ---------------- GEMM1: H = silu(Xg Gate^T) * (Xg Up^T), gathered rows ----------------
// 256 thr / 8 waves; tile 128M x 64N, K-step 32 over D. Double-buffered LDS:
// A tile via global_load_async_to_lds_b128 (bf16, no conversion), gate/up weights
// staged in registers and committed as packed bf16 overlapping the WMMAs.
__global__ __launch_bounds__(256) void moe_gemm1(
    const bf16* __restrict__ xb, const float* __restrict__ gate_w,
    const float* __restrict__ up_w, const int* __restrict__ lists,
    const unsigned* __restrict__ counts, bf16* __restrict__ H, int e)
{
  __shared__ bf16 sA[2][128 * LDK];
  __shared__ bf16 sG[2][64 * LDK];
  __shared__ bf16 sU[2][64 * LDK];
  __shared__ int sRow[128];
  int cnt = (int)counts[e];
  int m0 = blockIdx.y * 128;
  if (m0 >= cnt) return;
  int n0 = blockIdx.x * 64;
  const float* gw = gate_w + (size_t)e * FDIM * DDIM;
  const float* uw = up_w + (size_t)e * FDIM * DDIM;
  const int* lst = lists + (size_t)e * NTOK;
  int t = threadIdx.x;
  if (t < 128) {
    int i = m0 + t;
    sRow[t] = lst[i < cnt ? i : cnt - 1];
  }
  __syncthreads();
  int wave = t >> 5, lane = t & 31, lrow = lane & 15, half = lane >> 4;

  const int arow = t >> 1, aseg = t & 1;    // A: 128 rows x (2 x 16 halves)
  const int brow = t >> 2, bpart = t & 3;   // B: 64 rows x (4 x 8 floats)
  const bf16*  aptr = xb + (size_t)sRow[arow] * DDIM + aseg * 16;
  const float* gptr = gw + (size_t)(n0 + brow) * DDIM + bpart * 8;
  const float* uptr = uw + (size_t)(n0 + brow) * DDIM + bpart * 8;
  const unsigned ldsA[2] = { lds_off_of(&sA[0][arow * LDK + aseg * 16]),
                             lds_off_of(&sA[1][arow * LDK + aseg * 16]) };

  float4 rg0, rg1, ru0, ru1;
  auto stage = [&](int k0, int buf) {
    async_ld_b128x2(ldsA[buf], aptr + k0);          // A: global -> LDS, ASYNCcnt
    const float4* gs = (const float4*)(gptr + k0);
    rg0 = gs[0]; rg1 = gs[1];
    const float4* us = (const float4*)(uptr + k0);
    ru0 = us[0]; ru1 = us[1];
    __builtin_prefetch(gptr + k0 + 64, 0, 1);       // global_prefetch_b8, k+2 tile
    __builtin_prefetch(uptr + k0 + 64, 0, 1);
  };
  auto commit = [&](int buf) {
    *(uint4*)(&sG[buf][brow * LDK + bpart * 8]) = pack8(rg0, rg1);
    *(uint4*)(&sU[buf][brow * LDK + bpart * 8]) = pack8(ru0, ru1);
    wait_async0();                                  // A tile landed in LDS
  };

  v8f zf;
#pragma unroll
  for (int i = 0; i < 8; ++i) zf[i] = 0.f;
  v8f accg[4], accu[4];
#pragma unroll
  for (int nt = 0; nt < 4; ++nt) { accg[nt] = zf; accu[nt] = zf; }

  stage(0, 0);
  commit(0);
  __syncthreads();
  int p = 0;
  for (int k0 = 0; k0 < DDIM; k0 += 32) {
    bool nxt = (k0 + 32 < DDIM);
    if (nxt) stage(k0 + 32, p ^ 1);                 // next-tile loads in flight
    v16bf af = load_frag(&sA[p][(wave * 16 + lrow) * LDK], half);
    // software-pipelined B fragments: distinct regs so ds loads overlap WMMA
    v16bf bg = load_frag(&sG[p][lrow * LDK], half);
    v16bf bu = load_frag(&sU[p][lrow * LDK], half);
#pragma unroll
    for (int nt = 0; nt < 4; ++nt) {
      v16bf bg2 = bg, bu2 = bu;
      if (nt < 3) {
        bg2 = load_frag(&sG[p][((nt + 1) * 16 + lrow) * LDK], half);
        bu2 = load_frag(&sU[p][((nt + 1) * 16 + lrow) * LDK], half);
      }
      accg[nt] = __builtin_amdgcn_wmma_f32_16x16x32_bf16(
          false, af, false, bg, (short)0, accg[nt], false, false);
      accu[nt] = __builtin_amdgcn_wmma_f32_16x16x32_bf16(
          false, af, false, bu, (short)0, accu[nt], false, false);
      bg = bg2; bu = bu2;
    }
    __syncthreads();
    if (nxt) {
      commit(p ^ 1);
      p ^= 1;
      __syncthreads();
    }
  }
  // epilogue: C layout VGPR r -> M = r + 8*half; lane -> N = lane&15
  int mbase = m0 + wave * 16 + 8 * half;
#pragma unroll
  for (int nt = 0; nt < 4; ++nt) {
    int n = n0 + nt * 16 + lrow;
#pragma unroll
    for (int r = 0; r < 8; ++r) {
      int m = mbase + r;
      if (m < cnt) {
        float g = accg[nt][r], u = accu[nt][r];
        float sig = __builtin_amdgcn_rcpf(1.f + __expf(-g));  // v_rcp_f32, no IEEE div chain
        H[(size_t)m * FDIM + n] = f2bf(g * sig * u);          // silu(g)*u
      }
    }
  }
}

// ---------------- GEMM2: out[token] += w_token * (H Down^T) ----------------
// 256 thr; tile 128M x 128N, K-step 32 over F. A (H, bf16) via async global->LDS,
// down weights register-staged + packed to bf16.
__global__ __launch_bounds__(256) void moe_gemm2(
    const bf16* __restrict__ H, const float* __restrict__ down_w,
    const int* __restrict__ lists, const unsigned* __restrict__ counts,
    const float* __restrict__ combine, float* __restrict__ out, int e)
{
  __shared__ bf16 sA[2][128 * LDK];
  __shared__ bf16 sB[2][128 * LDK];
  __shared__ int sTok[128];
  __shared__ float sW[128];
  int cnt = (int)counts[e];
  int m0 = blockIdx.y * 128;
  if (m0 >= cnt) return;
  int n0 = blockIdx.x * 128;
  const float* dw = down_w + (size_t)e * DDIM * FDIM;
  const int* lst = lists + (size_t)e * NTOK;
  int t = threadIdx.x;
  if (t < 128) {
    int i = m0 + t;
    if (i < cnt) {
      int tok = lst[i];
      sTok[t] = tok;
      sW[t] = combine[(size_t)tok * NEXP + e];
    } else { sTok[t] = 0; sW[t] = 0.f; }
  }
  __syncthreads();
  int wave = t >> 5, lane = t & 31, lrow = lane & 15, half = lane >> 4;

  const int arow = t >> 1, aseg = t & 1;    // both tiles: 128 rows x (2 x 16 elems)
  int ai = m0 + arow; if (ai >= cnt) ai = cnt - 1;
  const bf16*  aptr = H  + (size_t)ai * FDIM + aseg * 16;
  const float* bptr = dw + (size_t)(n0 + arow) * FDIM + aseg * 16;
  const unsigned ldsA[2] = { lds_off_of(&sA[0][arow * LDK + aseg * 16]),
                             lds_off_of(&sA[1][arow * LDK + aseg * 16]) };

  float4 rb0, rb1, rb2, rb3;
  auto stage = [&](int k0, int buf) {
    async_ld_b128x2(ldsA[buf], aptr + k0);          // A: global -> LDS, ASYNCcnt
    const float4* bs = (const float4*)(bptr + k0);
    rb0 = bs[0]; rb1 = bs[1]; rb2 = bs[2]; rb3 = bs[3];
    __builtin_prefetch(bptr + k0 + 64, 0, 1);
  };
  auto commit = [&](int buf) {
    uint4* bd = (uint4*)(&sB[buf][arow * LDK + aseg * 16]);
    bd[0] = pack8(rb0, rb1);
    bd[1] = pack8(rb2, rb3);
    wait_async0();
  };

  v8f zf;
#pragma unroll
  for (int i = 0; i < 8; ++i) zf[i] = 0.f;
  v8f acc[8];
#pragma unroll
  for (int nt = 0; nt < 8; ++nt) acc[nt] = zf;

  stage(0, 0);
  commit(0);
  __syncthreads();
  int p = 0;
  for (int k0 = 0; k0 < FDIM; k0 += 32) {
    bool nxt = (k0 + 32 < FDIM);
    if (nxt) stage(k0 + 32, p ^ 1);
    v16bf af = load_frag(&sA[p][(wave * 16 + lrow) * LDK], half);
    v16bf bcur = load_frag(&sB[p][lrow * LDK], half);
#pragma unroll
    for (int nt = 0; nt < 8; ++nt) {
      v16bf bnxt = bcur;
      if (nt < 7) bnxt = load_frag(&sB[p][((nt + 1) * 16 + lrow) * LDK], half);
      acc[nt] = __builtin_amdgcn_wmma_f32_16x16x32_bf16(
          false, af, false, bcur, (short)0, acc[nt], false, false);
      bcur = bnxt;
    }
    __syncthreads();
    if (nxt) {
      commit(p ^ 1);
      p ^= 1;
      __syncthreads();
    }
  }
  // epilogue: scatter to token rows; experts run sequentially so plain RMW is race-free
  int mloc = wave * 16 + 8 * half;
#pragma unroll
  for (int nt = 0; nt < 8; ++nt) {
    int n = n0 + nt * 16 + lrow;
#pragma unroll
    for (int r = 0; r < 8; ++r) {
      int ml = mloc + r;
      if (m0 + ml < cnt) {
        float* o = out + (size_t)sTok[ml] * DDIM + n;
        *o += sW[ml] * acc[nt][r];
      }
    }
  }
}

// ---------------- launch ----------------
extern "C" void kernel_launch(void* const* d_in, const int* in_sizes, int n_in,
                              void* d_out, int out_size, void* d_ws, size_t ws_size,
                              hipStream_t stream) {
  const float* x  = (const float*)d_in[0];
  const float* rw = (const float*)d_in[1];
  const float* gw = (const float*)d_in[2];
  const float* uw = (const float*)d_in[3];
  const float* dw = (const float*)d_in[4];
  float* out = (float*)d_out;

  char* ws = (char*)d_ws;
  size_t off = 0;
  auto take = [&](size_t bytes) -> char* {
    char* p = ws + off;
    off = (off + bytes + 255) & ~(size_t)255;
    return p;
  };
  bf16*     xb      = (bf16*)take((size_t)NTOK * DDIM * 2);      // 33.6 MB
  float*    combine = (float*)take((size_t)NTOK * NEXP * 4);     // 256 KB
  unsigned* counts  = (unsigned*)take(NEXP * 4);
  int*      lists   = (int*)take((size_t)NEXP * NTOK * 4);       // 256 KB
  float*    spart   = (float*)take((size_t)RBLK * NEXP * 4);     // 32 KB
  bf16*     Hbuf    = (bf16*)take((size_t)NTOK * FDIM * 2);      // 92.3 MB (reused per expert)

  int n4 = NTOK * DDIM / 4;
  zero_f4<<<(n4 + 255) / 256, 256, 0, stream>>>((float4*)out, n4);
  zero_u<<<1, 32, 0, stream>>>(counts, NEXP);
  cvt_x<<<(n4 + 255) / 256, 256, 0, stream>>>(x, xb, n4);
  router_kernel<<<RBLK, 256, 0, stream>>>(x, rw, combine, counts, lists, spart);
  aux_finalize<<<1, 32, 0, stream>>>(spart, counts, out + (size_t)NTOK * DDIM, RBLK);
  for (int e = 0; e < NEXP; ++e) {
    moe_gemm1<<<dim3(FDIM / 64, NTOK / 128), 256, 0, stream>>>(xb, gw, uw, lists, counts, Hbuf, e);
    moe_gemm2<<<dim3(DDIM / 128, NTOK / 128), 256, 0, stream>>>(Hbuf, dw, lists, counts, combine, out, e);
  }
}
